// LinearAttentionTransformer_46179488366617
// MI455X (gfx1250) — compile-verified, matrix operand order checked by reading, not run on hardware
//
#include <hip/hip_runtime.h>
#include <hip/hip_bf16.h>
#include <math.h>
#include <stdint.h>

// ---------------- model constants ----------------
constexpr int S_ = 2048, D_ = 512, H_ = 8, L_ = 2, V_ = 32000, DH_ = 64, DFF_ = 2048;
constexpr float LN_EPS_ = 1e-5f;
constexpr float EPS_ = 1e-6f;

typedef _Float16 half_t;
typedef __attribute__((ext_vector_type(16))) _Float16 v16h;
typedef __attribute__((ext_vector_type(8)))  _Float16 v8h;
typedef __attribute__((ext_vector_type(8)))  float    v8f;

#if __has_builtin(__builtin_amdgcn_tensor_load_to_lds) && \
    __has_builtin(__builtin_amdgcn_s_wait_tensorcnt)
#define HAVE_TDM 1
#else
#define HAVE_TDM 0
#endif

// D = A(16x32) * B(32x16) + C ; f16 in, f32 accum
__device__ __forceinline__ v8f wmma_f16(v16h a, v16h b, v8f c) {
  return __builtin_amdgcn_wmma_f32_16x16x32_f16(false, a, false, b, (short)0, c,
                                                false, false);
}

// Load one 16x32 K-major fragment (A layout; also used for B via Bt = N x K).
// Lane layout (ISA 7.12.2, 16-bit A 16x32): lanes 0-15 row M=lane, halves 0..7 = K 0..7,
// halves 8..15 = K 16..23; lanes 16-31 row M=lane-16, K offset +8.
__device__ __forceinline__ v16h load_frag_kmajor(const half_t* __restrict__ base, int ld) {
  const int lane = threadIdx.x & 31;
  const int row  = lane & 15;
  const int koff = (lane < 16) ? 0 : 8;
  const half_t* p = base + (size_t)row * ld + koff;
  v8h lo = *(const v8h*)(p);
  v8h hi = *(const v8h*)(p + 16);
  v16h r;
#pragma unroll
  for (int i = 0; i < 8; ++i) { r[i] = lo[i]; r[i + 8] = hi[i]; }
  return r;
}

// Store a 16x16 f32 tile (C/D layout: VGPR r -> M=r (lanes 0-15) / M=8+r (lanes 16-31))
__device__ __forceinline__ void store_tile_f32(float* __restrict__ C, int ldc, v8f acc) {
  const int lane = threadIdx.x & 31;
  const int n  = lane & 15;
  const int mb = (lane < 16) ? 0 : 8;
#pragma unroll
  for (int r = 0; r < 8; ++r) C[(size_t)(mb + r) * ldc + n] = acc[r];
}

__device__ __forceinline__ float gelu_exact(float x) {
  return 0.5f * x * (1.0f + erff(x * 0.70710678118654752440f));
}

#if HAVE_TDM
typedef unsigned int u32x4 __attribute__((ext_vector_type(4)));
typedef int i32x4 __attribute__((ext_vector_type(4)));
typedef int i32x8 __attribute__((ext_vector_type(8)));

// Issue one TDM DMA: 2D tile (tile_dim0 = 32 elems along K, tile_dim1 = 128 rows) of
// f16 data from global (row stride = ld elements) into LDS at byte offset lds_off.
// D# group layout per CDNA5 ISA.
__device__ __forceinline__ void tdm_load_b_tile(const half_t* gsrc, unsigned lds_off,
                                                int ld) {
  unsigned long long ga = (unsigned long long)(uintptr_t)gsrc;
  u32x4 g0 = {};
  g0[0] = 1u;                                              // count=1, user mode
  g0[1] = lds_off;                                         // lds_addr (bytes)
  g0[2] = (unsigned)ga;                                    // global_addr[31:0]
  g0[3] = (unsigned)((ga >> 32) & 0x01FFFFFFu) | (2u << 30);  // addr[56:32] | type=2
  const unsigned dim0 = (unsigned)ld;                      // tensor_dim0 (elements)
  const unsigned dim1 = 128u;                              // tensor_dim1 (rows)
  const unsigned long long str0 = (unsigned long long)ld;  // tensor_dim0_stride
  i32x8 g1 = {};
  g1[0] = 0x00010000;                                      // data_size=1 -> 2 bytes
  g1[1] = (int)((dim0 & 0xFFFFu) << 16);                   // tensor_dim0[15:0]
  g1[2] = (int)((dim0 >> 16) | ((dim1 & 0xFFFFu) << 16));  // dim0[31:16], dim1[15:0]
  g1[3] = (int)((dim1 >> 16) | (32u << 16));               // dim1[31:16], tile_dim0=32
  g1[4] = (int)128;                                        // tile_dim1=128, tile_dim2=0
  g1[5] = (int)(unsigned)(str0 & 0xFFFFFFFFu);             // dim0_stride[31:0]
  g1[6] = (int)(unsigned)((str0 >> 32) & 0xFFFFu);         // dim0_stride[47:32]
  g1[7] = 0;
  i32x4 z4 = {};
#if defined(__clang_major__) && (__clang_major__ >= 23)
  i32x8 z8 = {};
  __builtin_amdgcn_tensor_load_to_lds(g0, g1, z4, z4, z8, 0);
#else
  __builtin_amdgcn_tensor_load_to_lds(g0, g1, z4, z4, 0);
#endif
}
#endif  // HAVE_TDM

// ---------------- embedding + sinusoidal PE ----------------
__global__ void embed_pe_kernel(const int* __restrict__ ids,
                                const float* __restrict__ emb,
                                float* __restrict__ x) {
  const int s = blockIdx.x;
  const int tok = ids[s];
  const float c = -logf(10000.0f) / (float)D_;
  for (int d = threadIdx.x; d < D_; d += blockDim.x) {
    int i = d >> 1;
    float f = expf((float)(2 * i) * c);
    float ang = (float)s * f;
    float pe = (d & 1) ? cosf(ang) : sinf(ang);
    x[(size_t)s * D_ + d] = emb[(size_t)tok * D_ + d] + pe;
  }
}

// ---------------- layernorm (f32 in -> f16 out) ----------------
__global__ __launch_bounds__(256) void layernorm_f16_kernel(
    const float* __restrict__ x, const float* __restrict__ scale,
    const float* __restrict__ bias, half_t* __restrict__ out) {
  const int s = blockIdx.x;
  const float* row = x + (size_t)s * D_;
  float lsum = 0.f, lsq = 0.f;
  for (int d = threadIdx.x; d < D_; d += 256) {
    float v = row[d];
    lsum += v; lsq += v * v;
  }
  __shared__ float ssum[8], ssq[8];
#pragma unroll
  for (int off = 16; off > 0; off >>= 1) {
    lsum += __shfl_down(lsum, off, 32);
    lsq  += __shfl_down(lsq,  off, 32);
  }
  const int lane = threadIdx.x & 31, w = threadIdx.x >> 5;
  if (lane == 0) { ssum[w] = lsum; ssq[w] = lsq; }
  __syncthreads();
  if (threadIdx.x == 0) {
    float ts = 0.f, tq = 0.f;
    for (int i = 0; i < 8; ++i) { ts += ssum[i]; tq += ssq[i]; }
    ssum[0] = ts; ssq[0] = tq;
  }
  __syncthreads();
  const float mu = ssum[0] / (float)D_;
  const float var = ssq[0] / (float)D_ - mu * mu;
  const float rstd = rsqrtf(var + LN_EPS_);
  for (int d = threadIdx.x; d < D_; d += 256) {
    float v = (row[d] - mu) * rstd * scale[d] + bias[d];
    out[(size_t)s * D_ + d] = (half_t)v;
  }
}

// ---------------- f32 KxN -> f16 NxK transpose-convert ----------------
__global__ void transpose_f32_to_f16_kernel(const float* __restrict__ src,
                                            half_t* __restrict__ dst, int K, int N) {
  __shared__ float tile[32][33];
  const int k0 = blockIdx.y * 32, n0 = blockIdx.x * 32;
  const int tx = threadIdx.x, ty = threadIdx.y;  // 32 x 8
  for (int r = ty; r < 32; r += 8) {
    int k = k0 + r, n = n0 + tx;
    tile[r][tx] = (k < K && n < N) ? src[(size_t)k * N + n] : 0.f;
  }
  __syncthreads();
  for (int r = ty; r < 32; r += 8) {
    int n = n0 + r, k = k0 + tx;
    if (n < N && k < K) dst[(size_t)n * K + k] = (half_t)tile[tx][r];
  }
}

// ---------------- WMMA GEMM: C(MxN,f32) = A(MxK,f16) * Bt(NxK,f16)^T ----------------
// 256 threads = 8 waves; block tile 64x128; each wave a 32x32 register tile (2x2 WMMA
// accumulators). B block-tile (128x32 f16 = 8KB) staged in LDS, double buffered, filled
// by a TDM tensor_load_to_lds DMA issued by wave 0 (s_wait_tensorcnt sync) when the
// toolchain exposes it, else by cooperative b128 loads. Requires M%64==0, N%128==0,
// K%32==0 (all call sites satisfy this; no partial tiles so EXEC stays all-ones).
template <bool HAS_BIAS, bool GELU, bool RESID, bool OUT16, bool OUT32>
__global__ __launch_bounds__(256) void gemm_f16_kernel(
    const half_t* __restrict__ A, const half_t* __restrict__ Bt,
    const float* __restrict__ bias, const float* __restrict__ resid,
    float* __restrict__ C32, half_t* __restrict__ C16,
    int M, int N, int K, int lda, int ldb, int ldc) {
  __shared__ __align__(16) half_t sB[2][128][32];
  const int w = threadIdx.x >> 5;
  const int m0 = blockIdx.y * 64 + (w & 1) * 32;   // global M of wave tile
  const int nb = blockIdx.x * 128;                 // block N base
  const int nl = (w >> 1) * 32;                    // local N of wave tile in block
  const int KT = K / 32;
  const half_t* btile = Bt + (size_t)nb * ldb;

  v8f acc00 = {}, acc01 = {}, acc10 = {}, acc11 = {};

#if HAVE_TDM
  const unsigned ldsbuf[2] = {(unsigned)(uintptr_t)&sB[0][0][0],
                              (unsigned)(uintptr_t)&sB[1][0][0]};
  if (w == 0) tdm_load_b_tile(btile, ldsbuf[0], ldb);
#endif

  for (int kt = 0; kt < KT; ++kt) {
    const int buf = kt & 1;
    const int k0 = kt * 32;
#if HAVE_TDM
    if (w == 0) {
      if (kt + 1 < KT) {
        tdm_load_b_tile(btile + (size_t)(kt + 1) * 32, ldsbuf[(kt + 1) & 1], ldb);
        __builtin_amdgcn_s_wait_tensorcnt(1);  // tile kt landed; kt+1 in flight
      } else {
        __builtin_amdgcn_s_wait_tensorcnt(0);
      }
    }
    __syncthreads();
#else
    // cooperative staging: 128 rows x 32 halfs, 16B chunks
    for (int c = threadIdx.x; c < 128 * 4; c += 256) {
      const int row = c >> 2, part = c & 3;
      v8h tmp = *(const v8h*)(btile + (size_t)row * ldb + k0 + part * 8);
      *(v8h*)(&sB[buf][row][part * 8]) = tmp;
    }
    __syncthreads();
#endif
    v16h a0 = load_frag_kmajor(A + (size_t)m0 * lda + k0, lda);
    v16h a1 = load_frag_kmajor(A + (size_t)(m0 + 16) * lda + k0, lda);
    v16h b0 = load_frag_kmajor(&sB[buf][nl][0], 32);
    v16h b1 = load_frag_kmajor(&sB[buf][nl + 16][0], 32);
    acc00 = wmma_f16(a0, b0, acc00);
    acc01 = wmma_f16(a0, b1, acc01);
    acc10 = wmma_f16(a1, b0, acc10);
    acc11 = wmma_f16(a1, b1, acc11);
    __syncthreads();  // all reads of sB[buf] done before it is refilled
  }

  // epilogue: 2x2 tiles of 16x16
  const int lane = threadIdx.x & 31;
  const int ln = lane & 15;
  const int lm = (lane < 16) ? 0 : 8;
#pragma unroll
  for (int mi = 0; mi < 2; ++mi) {
#pragma unroll
    for (int ni = 0; ni < 2; ++ni) {
      v8f acc = (mi == 0) ? (ni == 0 ? acc00 : acc01) : (ni == 0 ? acc10 : acc11);
      const int n = nb + nl + ni * 16 + ln;
      const int mbase = m0 + mi * 16 + lm;
      const float bv = HAS_BIAS ? bias[n] : 0.f;
#pragma unroll
      for (int r = 0; r < 8; ++r) {
        float v = acc[r] + bv;
        if (GELU) v = gelu_exact(v);
        size_t idx = (size_t)(mbase + r) * ldc + n;
        if (RESID) v += resid[idx];
        if (OUT32) C32[idx] = v;
        if (OUT16) C16[idx] = (half_t)v;
      }
    }
  }
}

// ---------------- qkv split + feature map phi(x)=elu(x)+1 ----------------
__global__ void phi_split_kernel(const float* __restrict__ qkv,
                                 half_t* __restrict__ q16, half_t* __restrict__ k16,
                                 half_t* __restrict__ kT16, half_t* __restrict__ vT16) {
  const int s = blockIdx.x;
  for (int d = threadIdx.x; d < D_; d += blockDim.x) {
    const int h = d >> 6, j = d & 63;
    float qv = qkv[(size_t)s * (3 * D_) + d];
    float kv = qkv[(size_t)s * (3 * D_) + D_ + d];
    float vv = qkv[(size_t)s * (3 * D_) + 2 * D_ + d];
    float pq = (qv > 0.f) ? (qv + 1.f) : expf(qv);  // elu(x)+1
    float pk = (kv > 0.f) ? (kv + 1.f) : expf(kv);
    size_t hsd = ((size_t)h * S_ + s) * DH_ + j;
    size_t hds = ((size_t)h * DH_ + j) * S_ + s;
    q16[hsd]  = (half_t)pq;
    k16[hsd]  = (half_t)pk;
    kT16[hds] = (half_t)pk;
    vT16[hds] = (half_t)vv;
  }
}

// ---------------- chunked linear causal attention, one block per head ----------------
__global__ __launch_bounds__(256) void linear_attn_kernel(
    const half_t* __restrict__ q16, const half_t* __restrict__ k16,
    const half_t* __restrict__ kT16, const half_t* __restrict__ vT16,
    half_t* __restrict__ attn16) {
  const int h = blockIdx.x;
  const half_t* Q  = q16  + (size_t)h * S_ * DH_;
  const half_t* Kk = k16  + (size_t)h * S_ * DH_;
  const half_t* KT = kT16 + (size_t)h * DH_ * S_;
  const half_t* VT = vT16 + (size_t)h * DH_ * S_;

  __shared__ __align__(16) float  sA[64][64];        // scores fp32
  __shared__ __align__(16) half_t sA16[64][64];      // masked scores f16
  __shared__ __align__(16) float  sStateT[64][64];   // [j][i] fp32 running state
  __shared__ __align__(16) half_t sState16[64][64];  // f16 snapshot (pre-chunk)
  __shared__ float sKsum[64];
  __shared__ float sRowSum[64];
  __shared__ float sQdot[64];

  const int tid = threadIdx.x;
  const int w = tid >> 5;
  const int tm = w & 3;

  for (int i = tid; i < 64 * 64; i += 256) (&sStateT[0][0])[i] = 0.f;
  if (tid < 64) sKsum[tid] = 0.f;
  __syncthreads();

  for (int c = 0; c < S_ / 64; ++c) {
    const int s0 = c * 64;

    // phase 1: scores A[m][n] = sum_i phi_q[s0+m][i] * phi_k[s0+n][i]
#pragma unroll
    for (int t = 0; t < 2; ++t) {
      const int tn = (w >> 2) + 2 * t;
      const int m0 = tm * 16, n0 = tn * 16;
      v8f acc = {};
      const half_t* ap = Q  + (size_t)(s0 + m0) * DH_;
      const half_t* bp = Kk + (size_t)(s0 + n0) * DH_;
      for (int k0 = 0; k0 < DH_; k0 += 32)
        acc = wmma_f16(load_frag_kmajor(ap + k0, DH_),
                       load_frag_kmajor(bp + k0, DH_), acc);
      store_tile_f32(&sA[m0][n0], 64, acc);
    }
    __syncthreads();

    // phase 2: mask+convert scores, snapshot state to f16, row sums, q.ksum_pre
    for (int i = tid; i < 64 * 64; i += 256) {
      int r = i >> 6, cc = i & 63;
      sA16[r][cc] = (half_t)((cc <= r) ? sA[r][cc] : 0.f);
      sState16[r][cc] = (half_t)sStateT[r][cc];
    }
    if (tid < 64) {
      float rs = 0.f;
      for (int cc = 0; cc <= tid; ++cc) rs += sA[tid][cc];
      float qd = 0.f;
      const half_t* qrow = Q + (size_t)(s0 + tid) * DH_;
      for (int i2 = 0; i2 < DH_; ++i2) qd += (float)qrow[i2] * sKsum[i2];
      sRowSum[tid] = rs;
      sQdot[tid] = qd;
    }
    __syncthreads();

    // phase 3a: numer[m][j] = q@S_pre + A_masked@v ; divide by denom; store
#pragma unroll
    for (int t = 0; t < 2; ++t) {
      const int tn = (w >> 2) + 2 * t;
      const int m0 = tm * 16, n0 = tn * 16;
      v8f acc = {};
      const half_t* ap = Q + (size_t)(s0 + m0) * DH_;
      for (int k0 = 0; k0 < DH_; k0 += 32)  // inter: K over i
        acc = wmma_f16(load_frag_kmajor(ap + k0, DH_),
                       load_frag_kmajor(&sState16[n0][k0], 64), acc);
      for (int k0 = 0; k0 < 64; k0 += 32)   // intra: K over s'
        acc = wmma_f16(load_frag_kmajor(&sA16[m0][k0], 64),
                       load_frag_kmajor(VT + (size_t)n0 * S_ + s0 + k0, S_), acc);
      const int lane = tid & 31;
      const int nn = n0 + (lane & 15);
      const int mb = m0 + ((lane < 16) ? 0 : 8);
#pragma unroll
      for (int r = 0; r < 8; ++r) {
        const int m = mb + r;
        float denom = sRowSum[m] + sQdot[m] + EPS_;
        attn16[(size_t)(s0 + m) * D_ + h * DH_ + nn] = (half_t)(acc[r] / denom);
      }
    }

    // phase 3b: state update SstT[j][i] += sum_s v[s][j]*phi_k[s][i]
#pragma unroll
    for (int t = 0; t < 2; ++t) {
      const int tn = (w >> 2) + 2 * t;
      const int m0 = tm * 16, n0 = tn * 16;  // m0->j, n0->i
      v8f acc = {};
      for (int k0 = 0; k0 < 64; k0 += 32)
        acc = wmma_f16(load_frag_kmajor(VT + (size_t)m0 * S_ + s0 + k0, S_),
                       load_frag_kmajor(KT + (size_t)n0 * S_ + s0 + k0, S_), acc);
      const int lane = tid & 31;
      const int nn = n0 + (lane & 15);
      const int mb = m0 + ((lane < 16) ? 0 : 8);
#pragma unroll
      for (int r = 0; r < 8; ++r) sStateT[mb + r][nn] += acc[r];
    }
    if (tid < 64) {
      float a = 0.f;
      for (int s2 = 0; s2 < 64; ++s2) a += (float)Kk[(size_t)(s0 + s2) * DH_ + tid];
      sKsum[tid] += a;
    }
    __syncthreads();
  }
}

// ---------------- host orchestration ----------------
extern "C" void kernel_launch(void* const* d_in, const int* in_sizes, int n_in,
                              void* d_out, int out_size, void* d_ws, size_t ws_size,
                              hipStream_t stream) {
  const int*   input_ids = (const int*)d_in[0];
  const float* token_emb = (const float*)d_in[1];
  const float* ln1_scale = (const float*)d_in[2];
  const float* ln1_bias  = (const float*)d_in[3];
  const float* qkv_w     = (const float*)d_in[4];
  const float* out_w     = (const float*)d_in[5];
  const float* ln2_scale = (const float*)d_in[6];
  const float* ln2_bias  = (const float*)d_in[7];
  const float* mlp_w1    = (const float*)d_in[8];
  const float* mlp_b1    = (const float*)d_in[9];
  const float* mlp_w2    = (const float*)d_in[10];
  const float* mlp_b2    = (const float*)d_in[11];
  const float* lnf_scale = (const float*)d_in[12];
  const float* lnf_bias  = (const float*)d_in[13];
  const float* head_w    = (const float*)d_in[14];
  float* logits = (float*)d_out;

  char* ws = (char*)d_ws;
  size_t off = 0;
  auto carve = [&](size_t bytes) -> char* {
    char* p = ws + off;
    off += (bytes + 255) & ~(size_t)255;
    return p;
  };
  float*  x      = (float*)carve((size_t)S_ * D_ * 4);
  half_t* xn16   = (half_t*)carve((size_t)S_ * D_ * 2);
  float*  qkv32  = (float*)carve((size_t)S_ * 3 * D_ * 4);
  half_t* q16    = (half_t*)carve((size_t)H_ * S_ * DH_ * 2);
  half_t* k16    = (half_t*)carve((size_t)H_ * S_ * DH_ * 2);
  half_t* kT16   = (half_t*)carve((size_t)H_ * DH_ * S_ * 2);
  half_t* vT16   = (half_t*)carve((size_t)H_ * DH_ * S_ * 2);
  half_t* attn16 = (half_t*)carve((size_t)S_ * D_ * 2);
  half_t* h16    = (half_t*)carve((size_t)S_ * DFF_ * 2);
  half_t* qkvT   = (half_t*)carve((size_t)3 * D_ * D_ * 2);
  half_t* outT   = (half_t*)carve((size_t)D_ * D_ * 2);
  half_t* w1T    = (half_t*)carve((size_t)DFF_ * D_ * 2);
  half_t* w2T    = (half_t*)carve((size_t)D_ * DFF_ * 2);
  half_t* headT  = (half_t*)carve((size_t)V_ * D_ * 2);

  const dim3 blk256(256);
  const dim3 tblk(32, 8);

  // head weight transpose (once)
  transpose_f32_to_f16_kernel<<<dim3(V_ / 32, D_ / 32), tblk, 0, stream>>>(
      head_w, headT, D_, V_);

  // embedding + positional encoding
  embed_pe_kernel<<<dim3(S_), blk256, 0, stream>>>(input_ids, token_emb, x);

  for (int l = 0; l < L_; ++l) {
    const float* qkv_wl = qkv_w + (size_t)l * D_ * 3 * D_;
    const float* out_wl = out_w + (size_t)l * D_ * D_;
    const float* w1l = mlp_w1 + (size_t)l * D_ * DFF_;
    const float* b1l = mlp_b1 + (size_t)l * DFF_;
    const float* w2l = mlp_w2 + (size_t)l * DFF_ * D_;
    const float* b2l = mlp_b2 + (size_t)l * D_;

    transpose_f32_to_f16_kernel<<<dim3(3 * D_ / 32, D_ / 32), tblk, 0, stream>>>(
        qkv_wl, qkvT, D_, 3 * D_);
    transpose_f32_to_f16_kernel<<<dim3(D_ / 32, D_ / 32), tblk, 0, stream>>>(
        out_wl, outT, D_, D_);
    transpose_f32_to_f16_kernel<<<dim3(DFF_ / 32, D_ / 32), tblk, 0, stream>>>(
        w1l, w1T, D_, DFF_);
    transpose_f32_to_f16_kernel<<<dim3(D_ / 32, DFF_ / 32), tblk, 0, stream>>>(
        w2l, w2T, DFF_, D_);

    // ln1 -> xn16
    layernorm_f16_kernel<<<dim3(S_), blk256, 0, stream>>>(
        x, ln1_scale + (size_t)l * D_, ln1_bias + (size_t)l * D_, xn16);

    // qkv = xn16 @ qkv_w  (M=S, N=3D, K=D), fp32 out
    gemm_f16_kernel<false, false, false, false, true>
        <<<dim3(3 * D_ / 128, S_ / 64), blk256, 0, stream>>>(
            xn16, qkvT, nullptr, nullptr, qkv32, nullptr,
            S_, 3 * D_, D_, D_, D_, 3 * D_);

    // feature map + head split
    phi_split_kernel<<<dim3(S_), blk256, 0, stream>>>(qkv32, q16, k16, kT16, vT16);

    // chunked linear causal attention
    linear_attn_kernel<<<dim3(H_), blk256, 0, stream>>>(q16, k16, kT16, vT16, attn16);

    // x = x + attn @ out_w  (residual fused)
    gemm_f16_kernel<false, false, true, false, true>
        <<<dim3(D_ / 128, S_ / 64), blk256, 0, stream>>>(
            attn16, outT, nullptr, x, x, nullptr,
            S_, D_, D_, D_, D_, D_);

    // ln2 -> xn16
    layernorm_f16_kernel<<<dim3(S_), blk256, 0, stream>>>(
        x, ln2_scale + (size_t)l * D_, ln2_bias + (size_t)l * D_, xn16);

    // h = gelu(xn16 @ w1 + b1), f16 out
    gemm_f16_kernel<true, true, false, true, false>
        <<<dim3(DFF_ / 128, S_ / 64), blk256, 0, stream>>>(
            xn16, w1T, b1l, nullptr, nullptr, h16,
            S_, DFF_, D_, D_, D_, DFF_);

    // x = x + h @ w2 + b2
    gemm_f16_kernel<true, false, true, false, true>
        <<<dim3(D_ / 128, S_ / 64), blk256, 0, stream>>>(
            h16, w2T, b2l, x, x, nullptr,
            S_, D_, DFF_, DFF_, DFF_, D_);
  }

  // final layernorm -> xn16
  layernorm_f16_kernel<<<dim3(S_), blk256, 0, stream>>>(x, lnf_scale, lnf_bias, xn16);

  // logits = xn16 @ head_w  (M=S, N=V, K=D) -> fp32 d_out
  gemm_f16_kernel<false, false, false, false, true>
      <<<dim3(V_ / 128, S_ / 64), blk256, 0, stream>>>(
          xn16, headT, nullptr, nullptr, logits, nullptr,
          S_, V_, D_, D_, D_, V_);
}
